// LinearCRF_27023934226780
// MI455X (gfx1250) — matrix operand order repaired and checked by reading, not compiled, not run on hardware
//
#include <hip/hip_runtime.h>
#include <hip/hip_bf16.h>

#define T_DIM 512
#define B_DIM 64
#define H_DIM 512
#define K_DIM 16

typedef __attribute__((ext_vector_type(16))) __bf16 v16bf;
typedef __attribute__((ext_vector_type(8)))  float  v8f;

// ---------- WMMA fragment loaders (CDNA5 16x16x32 bf16 layouts, ISA 7.12.2) ----------

// A fragment 16x32 bf16 from f32 global rows. row_ptr = &src[row * ld], row = m0 + (lane&15).
// lanes 0-15 hold K {0..7,16..23}; lanes 16-31 hold K {8..15,24..31}.
static __device__ __forceinline__ v16bf load_a_frag_f32(const float* __restrict__ row_ptr, int kbase) {
  const int half = (threadIdx.x & 31) >> 4;
  v16bf a;
#pragma unroll
  for (int v = 0; v < 8; ++v) {
    const int k0 = (v < 4) ? (2 * v + 8 * half) : (16 + 2 * (v - 4) + 8 * half);
    const float2 f = *(const float2*)(row_ptr + kbase + k0);
    a[2 * v]     = (__bf16)f.x;
    a[2 * v + 1] = (__bf16)f.y;
  }
  return a;
}

// A fragment 16x32 bf16 from LDS stage laid out row-major [16][32].
static __device__ __forceinline__ v16bf load_a_frag_lds(const __bf16* stage) {
  const int l = threadIdx.x & 31;
  const int half = l >> 4;
  const __bf16* rp = stage + (l & 15) * 32;
  v16bf a;
#pragma unroll
  for (int v = 0; v < 8; ++v) {
    const int k0 = (v < 4) ? (2 * v + 8 * half) : (16 + 2 * (v - 4) + 8 * half);
    a[2 * v]     = rp[k0];
    a[2 * v + 1] = rp[k0 + 1];
  }
  return a;
}

// B fragment 32x16 bf16 from row-major f32 W (ld cols). Lane = N column (n0+(lane&15)),
// lanes 0-15 hold K {0..7,16..23}; lanes 16-31 hold K {8..15,24..31}.
static __device__ __forceinline__ v16bf load_b_frag_f32(const float* __restrict__ W,
                                                        int kbase, int n0, int ld) {
  const int l = threadIdx.x & 31;
  const int half = l >> 4;
  const float* rp = W + n0 + (l & 15);
  v16bf b;
#pragma unroll
  for (int v = 0; v < 8; ++v) {
    const int k0 = (v < 4) ? (2 * v + 8 * half) : (16 + 2 * (v - 4) + 8 * half);
    b[2 * v]     = (__bf16)rp[(size_t)(kbase + k0) * ld];
    b[2 * v + 1] = (__bf16)rp[(size_t)(kbase + k0 + 1) * ld];
  }
  return b;
}

// ---------- Kernel 1: em = relu(hidden @ W1 + b1) @ W2 + b2, fused, WMMA bf16 ----------
// One wave per 16-row M-tile of the flattened (T*B, H) problem. The 16x512 ReLU'd h
// stripe never leaves the WGP: computed 32 columns at a time, staged to LDS as bf16,
// consumed immediately as the A operand of the @W2 WMMA.
__global__ __launch_bounds__(256) void crf_emissions_wmma(
    const float* __restrict__ hidden, const float* __restrict__ W1,
    const float* __restrict__ b1, const float* __restrict__ W2,
    const float* __restrict__ b2, float* __restrict__ em) {
  __shared__ __bf16 stage[8][16 * 32];   // 1 KB per wave
  const int wave = threadIdx.x >> 5;
  const int l = threadIdx.x & 31;
  const int half = l >> 4;
  const int m0 = (blockIdx.x * 8 + wave) * 16;
  const float* arow = hidden + (size_t)(m0 + (l & 15)) * H_DIM;
  __bf16* st = stage[wave];

  v8f em_acc = {0.f, 0.f, 0.f, 0.f, 0.f, 0.f, 0.f, 0.f};

  for (int p = 0; p < 16; ++p) {          // 32 columns of h per pass
#pragma unroll
    for (int sub = 0; sub < 2; ++sub) {
      const int n0 = p * 32 + sub * 16;
      v8f hC = {0.f, 0.f, 0.f, 0.f, 0.f, 0.f, 0.f, 0.f};
      for (int kc = 0; kc < H_DIM; kc += 32) {
        const v16bf af = load_a_frag_f32(arow, kc);
        const v16bf bf = load_b_frag_f32(W1, kc, n0, H_DIM);
        hC = __builtin_amdgcn_wmma_f32_16x16x32_bf16(false, af, false, bf,
                                                     (short)0, hC, false, false);
      }
      // bias + relu, stage to LDS as bf16 (C layout: N = lane&15, M = r + 8*half)
      const float bias = b1[n0 + (l & 15)];
      __bf16* wp = st + (half * 8) * 32 + sub * 16 + (l & 15);
#pragma unroll
      for (int r = 0; r < 8; ++r) {
        const float hv = fmaxf(hC[r] + bias, 0.f);
        wp[r * 32] = (__bf16)hv;
      }
    }
    // same-wave DS ops are in-order; wait + compiler fence before cross-lane read
    asm volatile("s_wait_dscnt 0x0" ::: "memory");
    const v16bf a2 = load_a_frag_lds(st);
    const v16bf b2f = load_b_frag_f32(W2, p * 32, 0, K_DIM);
    em_acc = __builtin_amdgcn_wmma_f32_16x16x32_bf16(false, a2, false, b2f,
                                                     (short)0, em_acc, false, false);
    asm volatile("" ::: "memory");        // keep next pass's stores after these loads
  }

  const float biask = b2[l & 15];
#pragma unroll
  for (int r = 0; r < 8; ++r) {
    const int row = m0 + half * 8 + r;    // row = t*B + b  ->  em[row*K + n]
    em[(size_t)row * K_DIM + (l & 15)] = em_acc[r] + biask;
  }
}

// ---------- Kernel 2: sequential CRF forward scan, 2 batches per wave, no barriers ----------
__global__ __launch_bounds__(32) void crf_forward_scan(
    const float* __restrict__ em, const float* __restrict__ trans,
    const int* __restrict__ lens, float* __restrict__ fwd) {
  __shared__ float sh[32];
  const int l = threadIdx.x;
  const int g = l >> 4;                   // batch slot within wave
  const int j = l & 15;                   // CRF state
  const int b = blockIdx.x * 2 + g;

  float tr[16];
#pragma unroll
  for (int i = 0; i < 16; ++i) tr[i] = trans[j * K_DIM + i];

  const int len = lens[b];
  float alpha = em[(size_t)b * K_DIM + j];
  float last = alpha;                     // t==0 case (lens>=T/2 anyway)

  for (int t = 1; t < T_DIM; ++t) {
    sh[l] = alpha;
    asm volatile("s_wait_dscnt 0x0" ::: "memory");   // intra-wave exchange, DS in-order
    float v[16];
#pragma unroll
    for (int i = 0; i < 16; ++i) v[i] = sh[g * 16 + i] + tr[i];
    float m = v[0];
#pragma unroll
    for (int i = 1; i < 16; ++i) m = fmaxf(m, v[i]);
    float s = 0.f;
#pragma unroll
    for (int i = 0; i < 16; ++i) s += __expf(v[i] - m);
    alpha = m + __logf(s) + em[((size_t)t * B_DIM + b) * K_DIM + j];
    if (t == len - 1) last = alpha;
    asm volatile("" ::: "memory");        // order next store after this pass's loads
  }

  sh[l] = last;
  asm volatile("s_wait_dscnt 0x0" ::: "memory");
  if (j == 0) {
    float m = sh[g * 16];
#pragma unroll
    for (int i = 1; i < 16; ++i) m = fmaxf(m, sh[g * 16 + i]);
    float s = 0.f;
#pragma unroll
    for (int i = 0; i < 16; ++i) s += __expf(sh[g * 16 + i] - m);
    fwd[b] = m + __logf(s);
  }
}

// ---------- Kernel 3: gold score. cumsum+gather == masked sum over t < len ----------
__global__ __launch_bounds__(32) void crf_gold(
    const float* __restrict__ em, const float* __restrict__ trans,
    const int* __restrict__ lens, const int* __restrict__ tags,
    float* __restrict__ gold) {
  const int b = blockIdx.x;
  const int l = threadIdx.x;
  const int len = lens[b];
  float s = 0.f;
  for (int t = l; t < len; t += 32) {
    const int tg = tags[t * B_DIM + b];
    float v = em[((size_t)t * B_DIM + b) * K_DIM + tg];
    if (t > 0) {
      const int tp = tags[(t - 1) * B_DIM + b];
      v += trans[tg * K_DIM + tp];
    }
    s += v;
  }
#pragma unroll
  for (int off = 16; off > 0; off >>= 1) s += __shfl_xor(s, off, 32);
  if (l == 0) gold[b] = s;
}

// ---------- Kernel 4: out = sum_b (fwd - gold) ----------
__global__ __launch_bounds__(32) void crf_final(
    const float* __restrict__ fwd, const float* __restrict__ gold,
    float* __restrict__ out) {
  const int l = threadIdx.x;
  float s = (fwd[l] - gold[l]) + (fwd[l + 32] - gold[l + 32]);
#pragma unroll
  for (int off = 16; off > 0; off >>= 1) s += __shfl_xor(s, off, 32);
  if (l == 0) out[0] = s;
}

extern "C" void kernel_launch(void* const* d_in, const int* in_sizes, int n_in,
                              void* d_out, int out_size, void* d_ws, size_t ws_size,
                              hipStream_t stream) {
  const float* hidden = (const float*)d_in[0];
  const float* W1     = (const float*)d_in[1];
  const float* b1     = (const float*)d_in[2];
  const float* W2     = (const float*)d_in[3];
  const float* b2     = (const float*)d_in[4];
  const float* trans  = (const float*)d_in[5];
  const int*   lens   = (const int*)d_in[6];
  const int*   tags   = (const int*)d_in[7];

  float* em   = (float*)d_ws;                                  // T*B*K f32 = 2 MB
  float* fwd  = em + (size_t)T_DIM * B_DIM * K_DIM;            // 64 f32
  float* gold = fwd + B_DIM;                                   // 64 f32
  float* out  = (float*)d_out;

  // 2048 M-tiles of 16 rows, one per wave, 8 waves per block
  crf_emissions_wmma<<<256, 256, 0, stream>>>(hidden, W1, b1, W2, b2, em);
  crf_forward_scan<<<32, 32, 0, stream>>>(em, trans, lens, fwd);
  crf_gold<<<64, 32, 0, stream>>>(em, trans, lens, tags, gold);
  crf_final<<<1, 32, 0, stream>>>(fwd, gold, out);
}
